// WheelMultiheadAttention_30339648979691
// MI455X (gfx1250) — compile-verified, matrix operand order checked by reading, not run on hardware
//
#include <hip/hip_runtime.h>

// ---------------------------------------------------------------------------
// WheelMultiheadAttention for MI455X (gfx1250, wave32, WMMA bf16)
// B=2, T=2048, C=1024, H=16, D=64
// d_out = [out (B*T*C fp32) | weights (B*H*T*T fp32)]
// ---------------------------------------------------------------------------

typedef __attribute__((ext_vector_type(16))) __bf16 bf16x16;
typedef __attribute__((ext_vector_type(8)))  __bf16 bf16x8;
typedef __attribute__((ext_vector_type(4)))  __bf16 bf16x4;
typedef __attribute__((ext_vector_type(16))) float  f32x16;
typedef __attribute__((ext_vector_type(8)))  float  f32x8;
typedef __attribute__((ext_vector_type(4)))  float  f32x4;

constexpr int Bc = 2, Tc = 2048, Cc = 1024, Hc = 16, Dc = 64;
constexpr int BT = Bc * Tc;

// Build a 16x32 bf16 A/B fragment from a bf16 row pointer.
// CDNA5 16-bit fragment striping: lanes 0-15 hold k in {0..7, 16..23},
// lanes 16-31 hold k in {8..15, 24..31} (two contiguous 16B chunks per lane).
__device__ inline bf16x16 load_frag(const __bf16* __restrict__ rowp, int kbase, int lane) {
  const int off = kbase + ((lane & 16) ? 8 : 0);
  const bf16x8 lo = *(const bf16x8*)(rowp + off);
  const bf16x8 hi = *(const bf16x8*)(rowp + off + 16);
  bf16x16 r;
#pragma unroll
  for (int i = 0; i < 8; ++i) { r[i] = lo[i]; r[i + 8] = hi[i]; }
  return r;
}

__device__ inline f32x8 wmma_bf16(bf16x16 a, bf16x16 b, f32x8 c) {
  return __builtin_amdgcn_wmma_f32_16x16x32_bf16(false, a, false, b, (short)0, c,
                                                 false, false);
}

// ---------------------------------------------------------------------------
// fp32 -> bf16 conversion, float4-vectorized (packed v_cvt_pk_bf16_f32)
// ---------------------------------------------------------------------------
__global__ void cvt_f32_to_bf16(const float* __restrict__ src,
                                __bf16* __restrict__ dst, int n4) {
  const int i = blockIdx.x * blockDim.x + threadIdx.x;
  if (i < n4) {
    const f32x4 v = ((const f32x4*)src)[i];
    *(bf16x4*)(dst + (size_t)i * 4) = __builtin_convertvector(v, bf16x4);
  }
}

// ---------------------------------------------------------------------------
// Y[M,N] = Xbf[M,K] @ Wbf[N,K]^T + bias ; M=BT, N=K=C.
// 256 threads = 8 waves, 4(M) x 2(N); wave tile 64x64 (16 WMMA / 32-k step).
// B (weight) tile is staged to LDS with ASYNCcnt-tracked async copies,
// double-buffered, so the 4 M-waves share one global fetch of each W row.
// mode 0: store bf16, per-head layout  [B,H,T,D]        (Q, K)
// mode 1: store bf16, transposed head  [B,H,D,T]        (V)
// mode 2: store fp32 row-major [BT,C] to d_out          (o-proj)
// ---------------------------------------------------------------------------
__global__ __launch_bounds__(256) void gemm_bf16(
    const __bf16* __restrict__ X, const __bf16* __restrict__ W,
    const float* __restrict__ bias, void* __restrict__ dst, int mode) {
  __shared__ __bf16 Bs[2][128][32];  // double-buffered 8 KB weight tile

  const int tid = threadIdx.x, lane = tid & 31, w = tid >> 5;
  const int wm = w & 3, wn = w >> 2;
  const int m0 = blockIdx.x * 256 + wm * 64;
  const int nb = blockIdx.y * 128;          // workgroup N base
  const int n0 = nb + wn * 64;

  const __bf16* arow[4];
#pragma unroll
  for (int i = 0; i < 4; ++i)
    arow[i] = X + (size_t)(m0 + i * 16 + (lane & 15)) * Cc;

  // Async-stage the 128x32 bf16 weight tile for k-block `kb` into buffer `buf`.
  // 512 x 16B chunks, 2 per thread; per-lane LDS dest + global src addresses.
  auto stage = [&](int kb, int buf) {
#pragma unroll
    for (int j = 0; j < 2; ++j) {
      const int chunk = tid + 256 * j;      // 0..511
      const int row = chunk >> 2;           // 0..127
      const int coff = (chunk & 3) * 8;     // bf16 element offset in row
      const __bf16* g = W + (size_t)(nb + row) * Cc + kb + coff;
      const unsigned lds = (unsigned)(uintptr_t)&Bs[buf][row][coff];
      asm volatile("global_load_async_to_lds_b128 %0, %1, off"
                   :: "v"(lds), "v"(g) : "memory");
    }
  };

  // Read a B fragment (tile rows rowbase..rowbase+15, k 0..31) from LDS.
  auto load_bfrag = [&](int buf, int rowbase) -> bf16x16 {
    const int off = (lane & 16) ? 8 : 0;
    const __bf16* p = &Bs[buf][rowbase + (lane & 15)][0];
    const bf16x8 lo = *(const bf16x8*)(p + off);
    const bf16x8 hi = *(const bf16x8*)(p + off + 16);
    bf16x16 r;
#pragma unroll
    for (int i = 0; i < 8; ++i) { r[i] = lo[i]; r[i + 8] = hi[i]; }
    return r;
  };

  f32x8 acc[4][4];
#pragma unroll
  for (int i = 0; i < 4; ++i)
#pragma unroll
    for (int j = 0; j < 4; ++j) acc[i][j] = (f32x8)0.0f;

  stage(0, 0);
  asm volatile("s_wait_asynccnt 0x0" ::: "memory");
  __syncthreads();

  int s = 0;
  for (int kb = 0; kb < Cc; kb += 32, s ^= 1) {
    const bool more = (kb + 32 < Cc);
    if (more) stage(kb + 32, s ^ 1);  // prefetch next tile while computing
    if (kb + 64 < Cc) __builtin_prefetch(arow[0] + kb + 64, 0, 1);

    bf16x16 a[4], b[4];
#pragma unroll
    for (int i = 0; i < 4; ++i) a[i] = load_frag(arow[i], kb, lane);
#pragma unroll
    for (int j = 0; j < 4; ++j) b[j] = load_bfrag(s, wn * 64 + j * 16);
#pragma unroll
    for (int i = 0; i < 4; ++i)
#pragma unroll
      for (int j = 0; j < 4; ++j) acc[i][j] = wmma_bf16(a[i], b[j], acc[i][j]);

    if (more) asm volatile("s_wait_asynccnt 0x0" ::: "memory");
    __syncthreads();  // buffer s consumed by all waves; s^1 fully landed
  }

  // Epilogue. C/D layout: lane l, vgpr r -> row = r + 8*(l>=16), col = l%16.
  const int rsub = (lane & 16) ? 8 : 0;
#pragma unroll
  for (int i = 0; i < 4; ++i) {
#pragma unroll
    for (int j = 0; j < 4; ++j) {
      const int col = n0 + j * 16 + (lane & 15);
      const float bv = bias[col];
      const int h = col >> 6, d = col & 63;
#pragma unroll
      for (int r = 0; r < 8; ++r) {
        const int m = m0 + i * 16 + rsub + r;
        const float v = acc[i][j][r] + bv;
        if (mode == 2) {
          ((float*)dst)[(size_t)m * Cc + col] = v;
        } else {
          const int bb = m >> 11, t = m & (Tc - 1);
          const size_t bh = (size_t)(bb * Hc + h);
          if (mode == 0)
            ((__bf16*)dst)[(bh * Tc + t) * Dc + d] = (__bf16)v;
          else
            ((__bf16*)dst)[(bh * Dc + d) * Tc + t] = (__bf16)v;
        }
      }
    }
  }
}

// ---------------------------------------------------------------------------
// Fused attention: one workgroup = (b, h, 16 query rows).
// Phase 1: S = scale*(Q K^T) with key-padding mask -> 128 KB LDS row-block.
// Phase 2: wheel softmax in LDS (float4-vectorized); stream normalized
//          weights to HBM exactly once with non-temporal float4 stores so
//          the 537 MB tensor never evicts the L2-resident Q/K/V (24 MB).
// Phase 3: out = W @ V straight out of LDS (packed fp32->bf16 fragments).
// ---------------------------------------------------------------------------
__global__ __launch_bounds__(256) void attn_fused(
    const __bf16* __restrict__ Qh,   // [B,H,T,D] bf16
    const __bf16* __restrict__ Kh,   // [B,H,T,D] bf16
    const __bf16* __restrict__ Vt,   // [B,H,D,T] bf16 (transposed)
    const unsigned char* __restrict__ mask,  // [B,T] bool
    float* __restrict__ Wout,        // [B,H,T,T] fp32 (d_out weights region)
    __bf16* __restrict__ AttnOut) {  // [B,T,C] bf16
  __shared__ float S[16][Tc];        // 128 KB score row-block
  __shared__ float redm[16][16];
  __shared__ float redz[16][16];
  __shared__ float Pout[2][16][Dc];  // K-split partials for W@V

  const int tid = threadIdx.x, lane = tid & 31, w = tid >> 5;
  const int bh = blockIdx.y;
  const int b = bh >> 4;             // H == 16
  const int q0 = blockIdx.x * 16;
  const float scale = 0.125f;        // D^-0.5
  const float NEG_INF = -__builtin_inff();
  const int rsub = (lane & 16) ? 8 : 0;

  // ---- Phase 1: scores (each wave owns 256 key columns) ----
  const __bf16* qrow = Qh + ((size_t)bh * Tc + q0 + (lane & 15)) * Dc;
  const bf16x16 qa0 = load_frag(qrow, 0, lane);
  const bf16x16 qa1 = load_frag(qrow, 32, lane);

  for (int cb = w * 256; cb < w * 256 + 256; cb += 16) {
    const __bf16* krow = Kh + ((size_t)bh * Tc + cb + (lane & 15)) * Dc;
    f32x8 acc = (f32x8)0.0f;
    acc = wmma_bf16(qa0, load_frag(krow, 0, lane), acc);
    acc = wmma_bf16(qa1, load_frag(krow, 32, lane), acc);
    const int col = cb + (lane & 15);
    const bool km = mask[b * Tc + col] != 0;
#pragma unroll
    for (int r = 0; r < 8; ++r)
      S[rsub + r][col] = km ? NEG_INF : acc[r] * scale;
  }
  __syncthreads();

  // ---- Phase 2: wheel softmax, float4-vectorized, lane-interleaved ----
  const int rr = tid >> 4, sg = tid & 15;   // 16 threads per row
  f32x4* Srow4 = (f32x4*)&S[rr][0];         // 512 float4 per row
  float mx = NEG_INF;
#pragma unroll 4
  for (int i = 0; i < 32; ++i) {
    const f32x4 v = Srow4[sg + 16 * i];
    mx = fmaxf(mx, fmaxf(fmaxf(v[0], v[1]), fmaxf(v[2], v[3])));
  }
  redm[rr][sg] = mx;
  __syncthreads();
  float rowm = NEG_INF;
#pragma unroll
  for (int i = 0; i < 16; ++i) rowm = fmaxf(rowm, redm[rr][i]);
  const float safem = (rowm == NEG_INF) ? 0.0f : rowm;  // all-masked row
  float z = 0.0f;
#pragma unroll 4
  for (int i = 0; i < 32; ++i) {
    const f32x4 v = Srow4[sg + 16 * i];
    f32x4 e;
    e[0] = __expf(v[0] - safem);            // exp(-inf) = 0
    e[1] = __expf(v[1] - safem);
    e[2] = __expf(v[2] - safem);
    e[3] = __expf(v[3] - safem);
    Srow4[sg + 16 * i] = e;
    z += e[0] + e[1] + e[2] + e[3];
  }
  redz[rr][sg] = z;
  __syncthreads();
  float rowz = 0.0f;
#pragma unroll
  for (int i = 0; i < 16; ++i) rowz += redz[rr][i];
  const float inv = (rowz > 0.0f) ? 1.0f / rowz : 0.0f;  // bot -> 0
  f32x4* wrow4 = (f32x4*)(Wout + ((size_t)bh * Tc + q0 + rr) * Tc);
#pragma unroll 4
  for (int i = 0; i < 32; ++i) {
    const int idx = sg + 16 * i;
    const f32x4 wv = Srow4[idx] * inv;
    Srow4[idx] = wv;
    __builtin_nontemporal_store(wv, &wrow4[idx]);  // single NT pass over 537 MB
  }
  __syncthreads();

  // ---- Phase 3: out = W @ V  (8 waves = 4 d-groups x 2 K-halves) ----
  const int ng = w & 3, kh = w >> 2;
  const __bf16* vrow = Vt + ((size_t)bh * Dc + ng * 16 + (lane & 15)) * Tc;
  const float* srow = &S[lane & 15][0];
  f32x8 acc = (f32x8)0.0f;
  for (int ks = 0; ks < 32; ++ks) {
    const int kbase = kh * 1024 + ks * 32;
    const int aoff = kbase + ((lane & 16) ? 8 : 0);
    const f32x4 a0 = *(const f32x4*)(srow + aoff);
    const f32x4 a1 = *(const f32x4*)(srow + aoff + 4);
    const f32x4 a2 = *(const f32x4*)(srow + aoff + 16);
    const f32x4 a3 = *(const f32x4*)(srow + aoff + 20);
    f32x16 af;
#pragma unroll
    for (int i = 0; i < 4; ++i) {
      af[i] = a0[i]; af[i + 4] = a1[i]; af[i + 8] = a2[i]; af[i + 12] = a3[i];
    }
    const bf16x16 a = __builtin_convertvector(af, bf16x16);  // packed cvt
    acc = wmma_bf16(a, load_frag(vrow, kbase, lane), acc);
  }
  {
    const int col = ng * 16 + (lane & 15);
#pragma unroll
    for (int r = 0; r < 8; ++r) Pout[kh][rsub + r][col] = acc[r];
  }
  __syncthreads();
  const int h = bh & 15;
  for (int i = tid; i < 16 * Dc; i += 256) {
    const int rq = i >> 6, d = i & 63;
    const float v = Pout[0][rq][d] + Pout[1][rq][d];
    AttnOut[((size_t)b * Tc + q0 + rq) * Cc + h * Dc + d] = (__bf16)v;
  }
}

// ---------------------------------------------------------------------------
extern "C" void kernel_launch(void* const* d_in, const int* in_sizes, int n_in,
                              void* d_out, int out_size, void* d_ws, size_t ws_size,
                              hipStream_t stream) {
  const float* query = (const float*)d_in[0];
  const float* key   = (const float*)d_in[1];
  const float* value = (const float*)d_in[2];
  const float* q_w = (const float*)d_in[3];
  const float* q_b = (const float*)d_in[4];
  const float* k_w = (const float*)d_in[5];
  const float* k_b = (const float*)d_in[6];
  const float* v_w = (const float*)d_in[7];
  const float* v_b = (const float*)d_in[8];
  const float* o_w = (const float*)d_in[9];
  const float* o_b = (const float*)d_in[10];
  const unsigned char* kpm = (const unsigned char*)d_in[11];

  char* ws = (char*)d_ws;
  size_t off = 0;
  auto alloc = [&](size_t bytes) {
    char* p = ws + off;
    off += (bytes + 255) & ~(size_t)255;
    return p;
  };
  __bf16* Xq = (__bf16*)alloc((size_t)BT * Cc * 2);
  __bf16* Xk = (__bf16*)alloc((size_t)BT * Cc * 2);
  __bf16* Xv = (__bf16*)alloc((size_t)BT * Cc * 2);
  __bf16* Wq = (__bf16*)alloc((size_t)Cc * Cc * 2);
  __bf16* Wk = (__bf16*)alloc((size_t)Cc * Cc * 2);
  __bf16* Wv = (__bf16*)alloc((size_t)Cc * Cc * 2);
  __bf16* Wo = (__bf16*)alloc((size_t)Cc * Cc * 2);
  __bf16* Qh = (__bf16*)alloc((size_t)Bc * Hc * Tc * Dc * 2);
  __bf16* Kh = (__bf16*)alloc((size_t)Bc * Hc * Tc * Dc * 2);
  __bf16* Vt = (__bf16*)alloc((size_t)Bc * Hc * Dc * Tc * 2);
  __bf16* AttnOut = (__bf16*)alloc((size_t)BT * Cc * 2);

  auto cvt = [&](const float* s, __bf16* d, int n) {
    const int n4 = n / 4;
    cvt_f32_to_bf16<<<(n4 + 255) / 256, 256, 0, stream>>>(s, d, n4);
  };
  cvt(query, Xq, BT * Cc);
  cvt(key,   Xk, BT * Cc);
  cvt(value, Xv, BT * Cc);
  cvt(q_w, Wq, Cc * Cc);
  cvt(k_w, Wk, Cc * Cc);
  cvt(v_w, Wv, Cc * Cc);
  cvt(o_w, Wo, Cc * Cc);

  const dim3 gg(BT / 256, Cc / 128);
  gemm_bf16<<<gg, 256, 0, stream>>>(Xq, Wq, q_b, (void*)Qh, 0);
  gemm_bf16<<<gg, 256, 0, stream>>>(Xk, Wk, k_b, (void*)Kh, 0);
  gemm_bf16<<<gg, 256, 0, stream>>>(Xv, Wv, v_b, (void*)Vt, 1);

  float* outp = (float*)d_out;
  float* Wout = outp + (size_t)BT * Cc;  // weights region after out
  const dim3 ga(Tc / 16, Bc * Hc);
  attn_fused<<<ga, 256, 0, stream>>>(Qh, Kh, Vt, kpm, Wout, AttnOut);

  gemm_bf16<<<gg, 256, 0, stream>>>(AttnOut, Wo, o_b, (void*)outp, 2);
}
